// Sinkhorn_wZatt_63891933496098
// MI455X (gfx1250) — matrix operand order repaired and checked by reading, not compiled, not run on hardware
//
#include <hip/hip_runtime.h>
#include <hip/hip_bf16.h>
#include <math.h>

typedef __attribute__((ext_vector_type(16))) _Float16 v16h;
typedef __attribute__((ext_vector_type(8)))  _Float16 v8h;
typedef __attribute__((ext_vector_type(8)))  float    v8f;

#define NORM_CONST (-7.6246190f)   /* -log(2048) */
#define LOG1024    (6.9314718f)    /* log(1024)  */

// ---------------------------------------------------------------------------
// f32 -> f16 conversion
// ---------------------------------------------------------------------------
__global__ __launch_bounds__(256) void k_f32_to_f16(const float* __restrict__ x,
                                                    _Float16* __restrict__ y, int n) {
  int t = blockIdx.x * blockDim.x + threadIdx.x;
  if (t < n) y[t] = (_Float16)x[t];
}

__device__ inline v16h make16(v8h lo, v8h hi) {
  v16h r;
#pragma unroll
  for (int t = 0; t < 8; ++t) { r[t] = lo[t]; r[8 + t] = hi[t]; }
  return r;
}

// ---------------------------------------------------------------------------
// LDS-staged, double-buffered WMMA f16 GEMM.
//   C[M,N] = scale*A*B (+biasM/+biasN)(relu)
// AMODE 0: A[m,k] = A[m*strideA + k]   (unit stride along K)
// AMODE 1: A[m,k] = A[k*strideA + m]   (unit stride along M)
// BMODE 0: B[k,n] = B[n*strideB + k]   (unit stride along K)
// BMODE 1: B[k,n] = B[k*strideB + n]   (unit stride along N)
// Block = 256 threads (8 waves) -> 128x64 C tile; wave = 32x32 (4 WMMA/chunk).
// Pipeline: store prefetched chunk to LDS[p], issue global loads for next
// chunk, one barrier, compute 4 WMMAs from LDS[p]. Fragment layouts per
// CDNA5 ISA 7.12.2 (wave32).
// ---------------------------------------------------------------------------
template<int AMODE, int BMODE>
__global__ __launch_bounds__(256) void k_wmma_gemm(
    const _Float16* __restrict__ A, int strideA,
    const _Float16* __restrict__ B, int strideB,
    float* __restrict__ Cf, _Float16* __restrict__ Ch, int ldc,
    int M, int N, int K,
    const float* __restrict__ biasM, const float* __restrict__ biasN,
    float scale, int relu)
{
  __shared__ _Float16 As[2][128 * 32];   // [m][k], k contiguous (64B rows)
  __shared__ _Float16 Bs[2][64 * 32];    // [n][k], k contiguous
  int tid  = threadIdx.x;
  int lane = tid & 31;
  int wave = tid >> 5;

  int nb  = N >> 6;
  int bm0 = (blockIdx.x / nb) << 7;
  int bn0 = (blockIdx.x % nb) << 6;

  int wm = (wave >> 1) << 5;        // 0,32,64,96
  int wn = (wave & 1) << 5;         // 0,32

  int aRow  = lane & 15;
  int aKoff = (lane >> 4) << 3;     // 0 / 8
  int bKoff = (lane >> 4) << 3;     // 0 / 8

  // ---- per-thread staging coordinates ----
  const _Float16* gA;
  int ldsA;
  if constexpr (AMODE == 0) {
    int m = tid >> 1, kq = (tid & 1) << 4;                 // 16 halves along K
    gA   = A + (size_t)(bm0 + m) * strideA + kq;
    ldsA = m * 32 + kq;
  } else {
    int k = tid >> 3, mq = (tid & 7) << 4;                 // 16 halves along M
    gA   = A + (size_t)k * strideA + (bm0 + mq);
    ldsA = mq * 32 + k;
  }
  const _Float16* gB;
  int ldsB;
  if constexpr (BMODE == 0) {
    int n = tid >> 2, kq = (tid & 3) << 3;                 // 8 halves along K
    gB   = B + (size_t)(bn0 + n) * strideB + kq;
    ldsB = n * 32 + kq;
  } else {
    int k = tid >> 3, nq = (tid & 7) << 3;                 // 8 halves along N
    gB   = B + (size_t)k * strideB + (bn0 + nq);
    ldsB = nq * 32 + k;
  }
  const size_t advA = (AMODE == 0) ? (size_t)32 : (size_t)32 * strideA;
  const size_t advB = (BMODE == 0) ? (size_t)32 : (size_t)32 * strideB;

  auto stageA = [&](int p, v8h r0, v8h r1) {
    if constexpr (AMODE == 0) {
      *(v8h*)(As[p] + ldsA)     = r0;
      *(v8h*)(As[p] + ldsA + 8) = r1;
    } else {
#pragma unroll
      for (int i = 0; i < 8; ++i) As[p][ldsA + i * 32]       = r0[i];
#pragma unroll
      for (int i = 0; i < 8; ++i) As[p][ldsA + (8 + i) * 32] = r1[i];
    }
  };
  auto stageB = [&](int p, v8h r) {
    if constexpr (BMODE == 0) {
      *(v8h*)(Bs[p] + ldsB) = r;
    } else {
#pragma unroll
      for (int i = 0; i < 8; ++i) Bs[p][ldsB + i * 32] = r[i];
    }
  };

  // ---- prefetch chunk 0 into registers ----
  v8h ra0 = *(const v8h*)gA;
  v8h ra1 = *(const v8h*)(gA + 8);
  v8h rb0 = *(const v8h*)gB;

  v8f acc00 = {}, acc01 = {}, acc10 = {}, acc11 = {};
  int p = 0;
  for (int k0 = 0; k0 < K; k0 += 32) {
    stageA(p, ra0, ra1);
    stageB(p, rb0);
    if (k0 + 32 < K) {                 // issue next-chunk loads before barrier
      gA += advA; gB += advB;
      ra0 = *(const v8h*)gA;
      ra1 = *(const v8h*)(gA + 8);
      rb0 = *(const v8h*)gB;
    }
    __syncthreads();

    const _Float16* pa0 = As[p] + (wm + aRow) * 32 + aKoff;
    const _Float16* pa1 = As[p] + (wm + 16 + aRow) * 32 + aKoff;
    const _Float16* pb0 = Bs[p] + (wn + (lane & 15)) * 32 + bKoff;
    const _Float16* pb1 = Bs[p] + (wn + 16 + (lane & 15)) * 32 + bKoff;
    v16h a0 = make16(*(const v8h*)pa0, *(const v8h*)(pa0 + 16));
    v16h a1 = make16(*(const v8h*)pa1, *(const v8h*)(pa1 + 16));
    v16h b0 = make16(*(const v8h*)pb0, *(const v8h*)(pb0 + 16));
    v16h b1 = make16(*(const v8h*)pb1, *(const v8h*)(pb1 + 16));

    acc00 = __builtin_amdgcn_wmma_f32_16x16x32_f16(false, a0, false, b0, (short)0, acc00, false, false);
    acc01 = __builtin_amdgcn_wmma_f32_16x16x32_f16(false, a0, false, b1, (short)0, acc01, false, false);
    acc10 = __builtin_amdgcn_wmma_f32_16x16x32_f16(false, a1, false, b0, (short)0, acc10, false, false);
    acc11 = __builtin_amdgcn_wmma_f32_16x16x32_f16(false, a1, false, b1, (short)0, acc11, false, false);
    p ^= 1;
  }

  // ---- epilogue: 4 16x16 tiles per wave ----
  int rbase = (lane >> 4) << 3;
  auto epi = [&](v8f acc, int mt, int nt) {
    int col = bn0 + wn + nt + (lane & 15);
    float bN = biasN ? biasN[col] : 0.f;
#pragma unroll
    for (int r = 0; r < 8; ++r) {
      int row = bm0 + wm + mt + rbase + r;
      float val = acc[r] * scale + bN + (biasM ? biasM[row] : 0.f);
      if (relu) val = fmaxf(val, 0.f);
      if (Cf) Cf[(size_t)row * ldc + col] = val;
      if (Ch) Ch[(size_t)row * ldc + col] = (_Float16)val;
    }
  };
  epi(acc00, 0, 0); epi(acc01, 0, 16); epi(acc10, 16, 0); epi(acc11, 16, 16);
}

// ---------------------------------------------------------------------------
// attn[i,j] = b2 + sum_c W2[c] * relu(hA[i,c] + hB[j,c])   (b1 folded into hA)
// ---------------------------------------------------------------------------
__global__ __launch_bounds__(256) void k_attn(const float* __restrict__ hA,
                                              const float* __restrict__ hB,
                                              const float* __restrict__ W2,
                                              const float* __restrict__ b2,
                                              float* __restrict__ attn) {
  __shared__ float sA[16 * 64];
  __shared__ float sB[16 * 64];
  __shared__ float sW[64];
  int tid = threadIdx.x;
  int i0 = blockIdx.y * 16, j0 = blockIdx.x * 16;
  for (int t = tid; t < 16 * 64; t += 256) {
    sA[t] = hA[(size_t)(i0 + (t >> 6)) * 64 + (t & 63)];
    sB[t] = hB[(size_t)(j0 + (t >> 6)) * 64 + (t & 63)];
  }
  if (tid < 64) sW[tid] = W2[tid];
  __syncthreads();
  int li = tid >> 4, lj = tid & 15;
  float acc = b2[0];
#pragma unroll
  for (int c = 0; c < 64; ++c)
    acc += sW[c] * fmaxf(sA[li * 64 + c] + sB[lj * 64 + c], 0.f);
  attn[(size_t)(i0 + li) * 1024 + (j0 + lj)] = acc;
}

// ---------------------------------------------------------------------------
// Generic softmax over n strided elements, f16 output (WMMA consumer).
// ---------------------------------------------------------------------------
__global__ __launch_bounds__(256) void k_softmax_f16(const float* __restrict__ X,
                                                     _Float16* __restrict__ P,
                                                     int n, int estride, int lstride) {
  const float* base = X + (size_t)blockIdx.x * lstride;
  _Float16*    out  = P + (size_t)blockIdx.x * lstride;
  __shared__ float red[256];
  float m = -1e30f;
  for (int e = threadIdx.x; e < n; e += 256) m = fmaxf(m, base[(size_t)e * estride]);
  red[threadIdx.x] = m; __syncthreads();
  for (int s = 128; s; s >>= 1) { if (threadIdx.x < s) red[threadIdx.x] = fmaxf(red[threadIdx.x], red[threadIdx.x + s]); __syncthreads(); }
  m = red[0]; __syncthreads();
  float sum = 0.f;
  for (int e = threadIdx.x; e < n; e += 256) sum += expf(base[(size_t)e * estride] - m);
  red[threadIdx.x] = sum; __syncthreads();
  for (int s = 128; s; s >>= 1) { if (threadIdx.x < s) red[threadIdx.x] += red[threadIdx.x + s]; __syncthreads(); }
  float inv = 1.f / red[0];
  for (int e = threadIdx.x; e < n; e += 256)
    out[(size_t)e * estride] = (_Float16)(expf(base[(size_t)e * estride] - m) * inv);
}

// ---------------------------------------------------------------------------
// zMLP final layer: z[r] = t2[r,:] . Wz3 + bz3
// ---------------------------------------------------------------------------
__global__ __launch_bounds__(256) void k_zfinal(const _Float16* __restrict__ t2,
                                                const float* __restrict__ Wz3,
                                                const float* __restrict__ bz3,
                                                float* __restrict__ z) {
  int r = blockIdx.x * blockDim.x + threadIdx.x;
  if (r >= 1024) return;
  float acc = bz3[0];
#pragma unroll
  for (int c = 0; c < 64; ++c) acc += (float)t2[(size_t)r * 64 + c] * Wz3[c];
  z[r] = acc;
}

// ---------------------------------------------------------------------------
// Dustbin row/col of couplings, u/v init
// ---------------------------------------------------------------------------
__global__ __launch_bounds__(256) void k_bins(float* __restrict__ C,
                                              const float* __restrict__ zA,
                                              const float* __restrict__ zB) {
  int t = blockIdx.x * blockDim.x + threadIdx.x;
  if (t < 1024) {
    C[(size_t)t * 1025 + 1024] = zA[t];
    C[(size_t)1024 * 1025 + t] = zB[t];
  }
  if (t == 1024) C[(size_t)1024 * 1025 + 1024] = 1.f;
}

__global__ __launch_bounds__(256) void k_init_uv(float* __restrict__ u, float* __restrict__ v) {
  int t = blockIdx.x * blockDim.x + threadIdx.x;
  if (t < 1025) { u[t] = 0.f; v[t] = 0.f; }
}

// ---------------------------------------------------------------------------
// Sinkhorn row pass: block per row (coalesced within row)
// ---------------------------------------------------------------------------
__global__ __launch_bounds__(256) void k_sink_row(const float* __restrict__ C,
                                                  const float* __restrict__ v,
                                                  float* __restrict__ u) {
  int i = blockIdx.x;
  const float* row = C + (size_t)i * 1025;
  __shared__ float red[256];
  float m = -1e30f;
  for (int j = threadIdx.x; j < 1025; j += 256) m = fmaxf(m, row[j] + v[j]);
  red[threadIdx.x] = m; __syncthreads();
  for (int s = 128; s; s >>= 1) { if (threadIdx.x < s) red[threadIdx.x] = fmaxf(red[threadIdx.x], red[threadIdx.x + s]); __syncthreads(); }
  m = red[0]; __syncthreads();
  float sum = 0.f;
  for (int j = threadIdx.x; j < 1025; j += 256) sum += expf(row[j] + v[j] - m);
  red[threadIdx.x] = sum; __syncthreads();
  for (int s = 128; s; s >>= 1) { if (threadIdx.x < s) red[threadIdx.x] += red[threadIdx.x + s]; __syncthreads(); }
  if (threadIdx.x == 0) {
    float lse = m + logf(red[0]);
    float lmu = (i < 1024) ? NORM_CONST : (LOG1024 + NORM_CONST);
    u[i] = lmu - lse;
  }
}

// ---------------------------------------------------------------------------
// Sinkhorn column pass: 32 columns x 8 row-slices per block (coalesced).
// ---------------------------------------------------------------------------
__global__ __launch_bounds__(256) void k_sink_col(const float* __restrict__ C,
                                                  const float* __restrict__ u,
                                                  float* __restrict__ v) {
  int cx = threadIdx.x & 31;
  int ry = threadIdx.x >> 5;             // 0..7
  int j = blockIdx.x * 32 + cx;
  bool active = (j < 1025);
  __shared__ float sm[8][33];
  __shared__ float ss[8][33];
  float m = -1e30f;
  if (active)
    for (int i = ry; i < 1025; i += 8) m = fmaxf(m, C[(size_t)i * 1025 + j] + u[i]);
  sm[ry][cx] = m; __syncthreads();
  float M = sm[0][cx];
#pragma unroll
  for (int r = 1; r < 8; ++r) M = fmaxf(M, sm[r][cx]);
  float s = 0.f;
  if (active)
    for (int i = ry; i < 1025; i += 8) s += expf(C[(size_t)i * 1025 + j] + u[i] - M);
  ss[ry][cx] = s; __syncthreads();
  if (ry == 0 && active) {
    float S = 0.f;
#pragma unroll
    for (int r = 0; r < 8; ++r) S += ss[r][cx];
    float lnu = (j < 1024) ? NORM_CONST : (LOG1024 + NORM_CONST);
    v[j] = lnu - (M + logf(S));
  }
}

// ---------------------------------------------------------------------------
// S[i,:] = softmax_j (C[i,j] + u[i] + v[j] - norm)
// ---------------------------------------------------------------------------
__global__ __launch_bounds__(256) void k_final_S(const float* __restrict__ C,
                                                 const float* __restrict__ u,
                                                 const float* __restrict__ v,
                                                 float* __restrict__ S) {
  int i = blockIdx.x;
  const float* row = C + (size_t)i * 1025;
  float ui = u[i] - NORM_CONST;
  __shared__ float red[256];
  float m = -1e30f;
  for (int j = threadIdx.x; j < 1025; j += 256) m = fmaxf(m, row[j] + ui + v[j]);
  red[threadIdx.x] = m; __syncthreads();
  for (int s = 128; s; s >>= 1) { if (threadIdx.x < s) red[threadIdx.x] = fmaxf(red[threadIdx.x], red[threadIdx.x + s]); __syncthreads(); }
  m = red[0]; __syncthreads();
  float sum = 0.f;
  for (int j = threadIdx.x; j < 1025; j += 256) sum += expf(row[j] + ui + v[j] - m);
  red[threadIdx.x] = sum; __syncthreads();
  for (int s = 128; s; s >>= 1) { if (threadIdx.x < s) red[threadIdx.x] += red[threadIdx.x + s]; __syncthreads(); }
  float inv = 1.f / red[0];
  for (int j = threadIdx.x; j < 1025; j += 256)
    S[(size_t)i * 1025 + j] = expf(row[j] + ui + v[j] - m) * inv;
}

// ---------------------------------------------------------------------------
// Argmax over inner S[0..1023,0..1023] (pitch 1025)
// ---------------------------------------------------------------------------
__global__ __launch_bounds__(256) void k_argmax_row(const float* __restrict__ S,
                                                    float* __restrict__ mx, int* __restrict__ ix) {
  int i = blockIdx.x;
  const float* row = S + (size_t)i * 1025;
  __shared__ float sv[256]; __shared__ int si[256];
  float bv = -1e30f; int bi = 0;
  for (int j = threadIdx.x; j < 1024; j += 256) { float x = row[j]; if (x > bv) { bv = x; bi = j; } }
  sv[threadIdx.x] = bv; si[threadIdx.x] = bi; __syncthreads();
  for (int s = 128; s; s >>= 1) {
    if (threadIdx.x < s) {
      float a = sv[threadIdx.x], b = sv[threadIdx.x + s];
      int ai = si[threadIdx.x], ci = si[threadIdx.x + s];
      if (b > a || (b == a && ci < ai)) { sv[threadIdx.x] = b; si[threadIdx.x] = ci; }
    }
    __syncthreads();
  }
  if (threadIdx.x == 0) { mx[i] = sv[0]; ix[i] = si[0]; }
}

__global__ __launch_bounds__(256) void k_argmax_col(const float* __restrict__ S,
                                                    float* __restrict__ mx, int* __restrict__ ix) {
  int cx = threadIdx.x & 31;
  int ry = threadIdx.x >> 5;
  int j = blockIdx.x * 32 + cx;
  bool active = (j < 1024);
  __shared__ float sv[8][33]; __shared__ int si[8][33];
  float bv = -1e30f; int bi = 1 << 30;
  if (active)
    for (int i = ry; i < 1024; i += 8) {
      float x = S[(size_t)i * 1025 + j];
      if (x > bv) { bv = x; bi = i; }
    }
  sv[ry][cx] = bv; si[ry][cx] = bi; __syncthreads();
  if (ry == 0 && active) {
    float V = sv[0][cx]; int I = si[0][cx];
#pragma unroll
    for (int r = 1; r < 8; ++r) {
      float v2 = sv[r][cx]; int i2 = si[r][cx];
      if (v2 > V || (v2 == V && i2 < I)) { V = v2; I = i2; }
    }
    mx[j] = V; ix[j] = I;
  }
}

// ---------------------------------------------------------------------------
// Mutual-NN matching + threshold. Output (floats):
//  [0,1024)=indices0  [1024,2048)=indices1  [2048,3072)=mscores0  [3072,4096)=mscores1
// ---------------------------------------------------------------------------
__global__ __launch_bounds__(256) void k_finalize(const int* __restrict__ idx0,
                                                  const float* __restrict__ max0,
                                                  const int* __restrict__ idx1,
                                                  float* __restrict__ out) {
  int t = blockIdx.x * blockDim.x + threadIdx.x;
  if (t >= 1024) return;
  int j0 = idx0[t];
  bool mut0 = (idx1[j0] == t);
  float ms0 = mut0 ? expf(max0[t]) : 0.f;
  bool val0 = mut0 && (ms0 > 0.2f);
  out[t]        = val0 ? (float)j0 : -1.f;   // indices0
  out[2048 + t] = ms0;                       // mscores0
  int i1 = idx1[t];
  bool mut1 = (idx0[i1] == t);
  bool mut0_i1 = (idx1[idx0[i1]] == i1);
  float ms0_i1 = mut0_i1 ? expf(max0[i1]) : 0.f;
  float ms1 = mut1 ? ms0_i1 : 0.f;
  bool val0_i1 = mut0_i1 && (ms0_i1 > 0.2f);
  bool val1 = mut1 && val0_i1;
  out[1024 + t] = val1 ? (float)i1 : -1.f;   // indices1
  out[3072 + t] = ms1;                       // mscores1
}

// ---------------------------------------------------------------------------
// Host launcher
// ---------------------------------------------------------------------------
extern "C" void kernel_launch(void* const* d_in, const int* in_sizes, int n_in,
                              void* d_out, int out_size, void* d_ws, size_t ws_size,
                              hipStream_t stream) {
  const float* desc0 = (const float*)d_in[0];   // 256x1024
  const float* desc1 = (const float*)d_in[1];   // 256x1024
  const float* Wp    = (const float*)d_in[2];   // 256x256
  const float* bp    = (const float*)d_in[3];   // 256
  const float* W1    = (const float*)d_in[4];   // 64x512
  const float* b1    = (const float*)d_in[5];   // 64
  const float* W2    = (const float*)d_in[6];   // 64
  const float* b2    = (const float*)d_in[7];   // 1
  const float* Wz1   = (const float*)d_in[8];   // 256x256
  const float* bz1   = (const float*)d_in[9];   // 256
  const float* Wz2   = (const float*)d_in[10];  // 64x256
  const float* bz2   = (const float*)d_in[11];  // 64
  const float* Wz3   = (const float*)d_in[12];  // 64
  const float* bz3   = (const float*)d_in[13];  // 1

  char* w = (char*)d_ws;
  auto alloc = [&](size_t bytes) -> char* {
    char* p = w; w += (bytes + 255) & ~(size_t)255; return p;
  };
  _Float16* Wp_h  = (_Float16*)alloc(65536 * 2);
  _Float16* d0_h  = (_Float16*)alloc(262144 * 2);
  _Float16* d1_h  = (_Float16*)alloc(262144 * 2);
  _Float16* W1_h  = (_Float16*)alloc(32768 * 2);
  _Float16* Wz1_h = (_Float16*)alloc(65536 * 2);
  _Float16* Wz2_h = (_Float16*)alloc(16384 * 2);
  _Float16* md0_h = (_Float16*)alloc(262144 * 2);
  _Float16* md1_h = (_Float16*)alloc(262144 * 2);
  _Float16* pA_h  = (_Float16*)alloc(1048576 * 2);
  _Float16* pB_h  = (_Float16*)alloc(1048576 * 2);
  _Float16* zfA_h = (_Float16*)alloc(262144 * 2);
  _Float16* zfB_h = (_Float16*)alloc(262144 * 2);
  _Float16* t1_h  = (_Float16*)alloc(262144 * 2);
  _Float16* t2_h  = (_Float16*)alloc(65536 * 2);
  float* hA   = (float*)alloc(65536 * 4);
  float* hB   = (float*)alloc(65536 * 4);
  float* attn = (float*)alloc(1048576 * 4);
  float* coup = (float*)alloc((size_t)1050625 * 4);
  float* u    = (float*)alloc(1025 * 4);
  float* v    = (float*)alloc(1025 * 4);
  float* zA   = (float*)alloc(1024 * 4);
  float* zB   = (float*)alloc(1024 * 4);
  float* max0 = (float*)alloc(1024 * 4);
  float* max1 = (float*)alloc(1024 * 4);
  int*   idx0 = (int*)alloc(1024 * 4);
  int*   idx1 = (int*)alloc(1024 * 4);

  auto conv = [&](const float* src, _Float16* dst, int n) {
    k_f32_to_f16<<<(n + 255) / 256, 256, 0, stream>>>(src, dst, n);
  };
  auto gemm = [&](const _Float16* A, int amode, int strideA,
                  const _Float16* B, int bmode, int strideB,
                  float* Cf, _Float16* Ch, int ldc,
                  int M, int N, int K,
                  const float* bM, const float* bN, float scale, int relu) {
    int blocks = (M / 128) * (N / 64);
    if (amode == 0 && bmode == 0)
      k_wmma_gemm<0,0><<<blocks, 256, 0, stream>>>(A, strideA, B, strideB, Cf, Ch, ldc, M, N, K, bM, bN, scale, relu);
    else if (amode == 0 && bmode == 1)
      k_wmma_gemm<0,1><<<blocks, 256, 0, stream>>>(A, strideA, B, strideB, Cf, Ch, ldc, M, N, K, bM, bN, scale, relu);
    else if (amode == 1 && bmode == 0)
      k_wmma_gemm<1,0><<<blocks, 256, 0, stream>>>(A, strideA, B, strideB, Cf, Ch, ldc, M, N, K, bM, bN, scale, relu);
    else
      k_wmma_gemm<1,1><<<blocks, 256, 0, stream>>>(A, strideA, B, strideB, Cf, Ch, ldc, M, N, K, bM, bN, scale, relu);
  };

  // convert weights + descriptors to f16
  conv(Wp, Wp_h, 65536);
  conv(desc0, d0_h, 262144);
  conv(desc1, d1_h, 262144);
  conv(W1, W1_h, 32768);
  conv(Wz1, Wz1_h, 65536);
  conv(Wz2, Wz2_h, 16384);

  // mdesc = Wp @ desc + bp  (f16 out).  A=Wp (unit-K), B=desc (unit-N)
  gemm(Wp_h, 0, 256, d0_h, 1, 1024, nullptr, md0_h, 1024, 256, 1024, 256, bp, nullptr, 1.f, 0);
  gemm(Wp_h, 0, 256, d1_h, 1, 1024, nullptr, md1_h, 1024, 256, 1024, 256, bp, nullptr, 1.f, 0);

  // hA[n,j] = sum_d mdesc0[d,n]*W1[j,d] + b1[j] ; hB without bias
  gemm(md0_h, 1, 1024, W1_h,       0, 512, hA, nullptr, 64, 1024, 64, 256, nullptr, b1,      1.f, 0);
  gemm(md1_h, 1, 1024, W1_h + 256, 0, 512, hB, nullptr, 64, 1024, 64, 256, nullptr, nullptr, 1.f, 0);

  // attention map + dual softmax
  k_attn<<<dim3(64, 64), 256, 0, stream>>>(hA, hB, W2, b2, attn);
  k_softmax_f16<<<1024, 256, 0, stream>>>(attn, pA_h, 1024, 1, 1024);  // rows (axis=1)
  k_softmax_f16<<<1024, 256, 0, stream>>>(attn, pB_h, 1024, 1024, 1);  // cols (axis=0)

  // zfeatA[i,d] = sum_j probA[i,j]*mdesc1[d,j] ; zfeatB[j,d] = sum_i probB[i,j]*mdesc0[d,i]
  gemm(pA_h, 0, 1024, md1_h, 0, 1024, nullptr, zfA_h, 256, 1024, 256, 1024, nullptr, nullptr, 1.f, 0);
  gemm(pB_h, 1, 1024, md0_h, 0, 1024, nullptr, zfB_h, 256, 1024, 256, 1024, nullptr, nullptr, 1.f, 0);

  // zMLP for A
  gemm(zfA_h, 0, 256, Wz1_h, 0, 256, nullptr, t1_h, 256, 1024, 256, 256, nullptr, bz1, 1.f, 1);
  gemm(t1_h,  0, 256, Wz2_h, 0, 256, nullptr, t2_h, 64,  1024, 64,  256, nullptr, bz2, 1.f, 1);
  k_zfinal<<<4, 256, 0, stream>>>(t2_h, Wz3, bz3, zA);
  // zMLP for B (reuse t1/t2)
  gemm(zfB_h, 0, 256, Wz1_h, 0, 256, nullptr, t1_h, 256, 1024, 256, 256, nullptr, bz1, 1.f, 1);
  gemm(t1_h,  0, 256, Wz2_h, 0, 256, nullptr, t2_h, 64,  1024, 64,  256, nullptr, bz2, 1.f, 1);
  k_zfinal<<<4, 256, 0, stream>>>(t2_h, Wz3, bz3, zB);

  // scores[n,m] = mdesc0[:,n].mdesc1[:,m] / 16 -> couplings inner block (pitch 1025)
  gemm(md0_h, 1, 1024, md1_h, 1, 1024, coup, nullptr, 1025, 1024, 1024, 256,
       nullptr, nullptr, 0.0625f, 0);
  k_bins<<<5, 256, 0, stream>>>(coup, zA, zB);
  k_init_uv<<<5, 256, 0, stream>>>(u, v);

  // 100 Sinkhorn iterations (u-update then v-update)
  for (int it = 0; it < 100; ++it) {
    k_sink_row<<<1025, 256, 0, stream>>>(coup, v, u);
    k_sink_col<<<33, 256, 0, stream>>>(coup, u, v);
  }

  float* out = (float*)d_out;
  float* S   = out + 4096;
  k_final_S<<<1025, 256, 0, stream>>>(coup, u, v, S);
  k_argmax_row<<<1024, 256, 0, stream>>>(S, max0, idx0);
  k_argmax_col<<<32, 256, 0, stream>>>(S, max1, idx1);
  k_finalize<<<4, 256, 0, stream>>>(idx0, max0, idx1, out);
  (void)in_sizes; (void)n_in; (void)out_size; (void)ws_size;
}
